// CausalMultiHeadSelfAttention_7103875907995
// MI455X (gfx1250) — matrix operand order repaired and checked by reading, not compiled, not run on hardware
//
#include <hip/hip_runtime.h>
#include <hip/hip_bf16.h>
#include <math.h>

// ---------------------------------------------------------------------------
// Causal MHA for MI455X (gfx1250, wave32, WMMA 16x16x32 f16->f32).
//   cvt f32->f16 | QKV GEMM (TDM-staged LDS-blocked WMMA; V stored transposed)
//   | RoPE | flash attention (32 queries/wave, all-contiguous operands)
//   | output GEMM (same blocking).
// ---------------------------------------------------------------------------

typedef _Float16     h8   __attribute__((ext_vector_type(8)));
typedef _Float16     v16h __attribute__((ext_vector_type(16)));
typedef float        v8f  __attribute__((ext_vector_type(8)));
typedef unsigned int v4u  __attribute__((ext_vector_type(4)));
typedef int          v4i  __attribute__((ext_vector_type(4)));
typedef int          v8i  __attribute__((ext_vector_type(8)));

#define D_MODEL 1024
#define N_HEADS 16
#define D_K     64
#define SEQ     4096

#if defined(__has_builtin)
#if __has_builtin(__builtin_amdgcn_tensor_load_to_lds)
#define HAVE_TDM 1
#endif
#endif
#ifndef HAVE_TDM
#define HAVE_TDM 0
#endif

static __device__ inline v16h pack16(h8 lo, h8 hi) {
  return __builtin_shufflevector(lo, hi, 0, 1, 2, 3, 4, 5, 6, 7,
                                         8, 9, 10, 11, 12, 13, 14, 15);
}

// A operand 16x32 f16 (ISA 7.12.2): lane<16 -> row=lane, K {0..7,16..23};
// lane>=16 -> row=lane-16, K {8..15,24..31}.
static __device__ inline v16h load_a(const _Float16* base, int ld, int kbase, int lane) {
  const int r  = lane & 15;
  const int k0 = kbase + ((lane & 16) >> 1);
  const _Float16* p = base + (size_t)r * ld + k0;
  return pack16(*(const h8*)p, *(const h8*)(p + 16));
}

// B operand 32x16 f16 where B[k][n] = src[n][k]: lane%16 -> column n;
// lanes 0-15 hold K {0..7,16..23}, lanes 16-31 hold K {8..15,24..31}.
static __device__ inline v16h load_b(const _Float16* base, int ld, int kbase, int lane) {
  const int c  = lane & 15;
  const int k0 = kbase + ((lane & 16) >> 1);
  const _Float16* p = base + (size_t)c * ld + k0;
  return pack16(*(const h8*)p, *(const h8*)(p + 16));
}

// Low 32 bits of a generic pointer to __shared__ = LDS byte offset.
static __device__ inline unsigned int lds_off(const void* p) {
  return (unsigned int)(uintptr_t)p;
}

#if HAVE_TDM
// One TDM 2D tile load: (tile0 x tile1) elements of 2-byte data from a
// row-major tensor (dim0 x dim1, stride0 elems) into LDS, rows packed.
static __device__ inline void tdm_load_2d(const _Float16* gptr, unsigned int ldsaddr,
                                          unsigned int dim0, unsigned int dim1,
                                          unsigned int stride0,
                                          unsigned int tile0, unsigned int tile1) {
  union { unsigned int u[4]; v4u v; } g0;
  union { unsigned int u[8]; v8i v; } g1;
  const unsigned long long ga = (unsigned long long)(uintptr_t)gptr;

  g0.u[0] = 1u;                                        // count=1 (valid, user)
  g0.u[1] = ldsaddr;                                   // lds_addr [63:32]
  g0.u[2] = (unsigned int)ga;                          // global_addr [95:64]
  g0.u[3] = (unsigned int)((ga >> 32) & 0x01FFFFFFu)   // global_addr [120:96]
          | (2u << 30);                                // type=2 ("image")

  for (int i = 0; i < 8; ++i) g1.u[i] = 0;
  g1.u[0]  = 1u << 16;                                 // data_size=1 (2 bytes)
  g1.u[1] |= (dim0 & 0xFFFFu) << 16;                   // tensor_dim0 [79:48]
  g1.u[2] |= (dim0 >> 16) & 0xFFFFu;
  g1.u[2] |= (dim1 & 0xFFFFu) << 16;                   // tensor_dim1 [111:80]
  g1.u[3] |= (dim1 >> 16) & 0xFFFFu;
  g1.u[3] |= (tile0 & 0xFFFFu) << 16;                  // tile_dim0 [127:112]
  g1.u[4] |= (tile1 & 0xFFFFu);                        // tile_dim1 [143:128]
  g1.u[5]  = stride0;                                  // tensor_dim0_stride [207:160]

  const v4i z4 = {0, 0, 0, 0};
#if defined(__clang_major__) && (__clang_major__ >= 23)
  const v8i z8 = {0, 0, 0, 0, 0, 0, 0, 0};
  __builtin_amdgcn_tensor_load_to_lds(g0.v, g1.v, z4, z4, z8, 0);
#else
  __builtin_amdgcn_tensor_load_to_lds(g0.v, g1.v, z4, z4, 0);
#endif
}
#endif  // HAVE_TDM

// ---------------------------------------------------------------------------
// 128x128-per-block GEMM mainloop; k-sliced by 32, TDM double-buffered LDS.
// Wave grid 4(M) x 2(N); each wave owns a 32x64 sub-tile (8 WMMAs / k-step).
static __device__ inline void gemm_tile_128x128(const _Float16* __restrict__ Ag,
                                                const _Float16* __restrict__ Bg,
                                                unsigned int a_rows, unsigned int b_rows,
                                                _Float16* Abuf, _Float16* Bbuf,
                                                v8f acc[2][4]) {
  const int lane = threadIdx.x & 31;
  const int lwid = threadIdx.x >> 5;
  const int wm   = lwid & 3;
  const int wn   = lwid >> 2;
  const int NK   = D_MODEL / 32;

#if HAVE_TDM
  if (lwid == 0) {
    tdm_load_2d(Ag, lds_off(Abuf), D_MODEL, a_rows, D_MODEL, 32, 128);
    tdm_load_2d(Bg, lds_off(Bbuf), D_MODEL, b_rows, D_MODEL, 32, 128);
  }
#endif

  for (int i = 0; i < NK; ++i) {
    const int cur = i & 1;
#if HAVE_TDM
    const int nxt = cur ^ 1;
    if (lwid == 0) {
      if (i + 1 < NK) {
        tdm_load_2d(Ag + (i + 1) * 32, lds_off(Abuf + nxt * 4096),
                    D_MODEL, a_rows, D_MODEL, 32, 128);
        tdm_load_2d(Bg + (i + 1) * 32, lds_off(Bbuf + nxt * 4096),
                    D_MODEL, b_rows, D_MODEL, 32, 128);
        __builtin_amdgcn_s_wait_tensorcnt(2);
      } else {
        __builtin_amdgcn_s_wait_tensorcnt(0);
      }
    }
#else
    {
      const int tid = threadIdx.x;
      #pragma unroll
      for (int t = 0; t < 2; ++t) {
        const int idx = (tid + t * 256) * 8;
        const int row = idx >> 5, col = idx & 31;
        *(h8*)(Abuf + cur * 4096 + idx) =
            *(const h8*)(Ag + (size_t)row * D_MODEL + i * 32 + col);
        *(h8*)(Bbuf + cur * 4096 + idx) =
            *(const h8*)(Bg + (size_t)row * D_MODEL + i * 32 + col);
      }
    }
#endif
    __syncthreads();

    const _Float16* Al = Abuf + cur * 4096;   // [128][32]
    const _Float16* Bl = Bbuf + cur * 4096;   // [128][32]
    const int r   = lane & 15;
    const int ak0 = (lane & 16) >> 1;
    const int bk0 = (lane & 16) >> 1;

    v16h af[2], bf[4];
    #pragma unroll
    for (int im = 0; im < 2; ++im) {
      const _Float16* p = Al + (wm * 32 + im * 16 + r) * 32 + ak0;
      af[im] = pack16(*(const h8*)p, *(const h8*)(p + 16));
    }
    #pragma unroll
    for (int jn = 0; jn < 4; ++jn) {
      const _Float16* p = Bl + (wn * 64 + jn * 16 + r) * 32 + bk0;
      bf[jn] = pack16(*(const h8*)p, *(const h8*)(p + 16));
    }
    #pragma unroll
    for (int im = 0; im < 2; ++im)
      #pragma unroll
      for (int jn = 0; jn < 4; ++jn)
        acc[im][jn] = __builtin_amdgcn_wmma_f32_16x16x32_f16(
            false, af[im], false, bf[jn], (short)0, acc[im][jn], false, false);

    __syncthreads();
  }
}

// ---------------------------------------------------------------------------
__global__ __launch_bounds__(256) void cvt_f32_f16(const float* __restrict__ src,
                                                   _Float16* __restrict__ dst, int n) {
  for (int i = blockIdx.x * blockDim.x + threadIdx.x; i < n; i += gridDim.x * blockDim.x)
    dst[i] = (_Float16)src[i];
}

// ---------------------------------------------------------------------------
// QKV projection: out[s,o] = sum_i x[s,i] * W[o,i].
// Q,K scatter into [h][s][64]; V stores TRANSPOSED into Vt[o][s] ([h][64][seq])
// with vectorized 8-half stores so attention's P.V B-operand is contiguous.
__global__ __launch_bounds__(256) void qkv_gemm(const _Float16* __restrict__ xh,
                                                const _Float16* __restrict__ wq,
                                                const _Float16* __restrict__ wk,
                                                const _Float16* __restrict__ wv,
                                                _Float16* __restrict__ Qh,
                                                _Float16* __restrict__ Kh,
                                                _Float16* __restrict__ Vt) {
  __shared__ _Float16 Abuf[2][128 * 32];
  __shared__ _Float16 Bbuf[2][128 * 32];

  const int bid = blockIdx.x;
  const int mat = bid >> 8;
  const int rem = bid & 255;
  const int mt  = rem >> 3;
  const int nt  = rem & 7;

  const _Float16* W = (mat == 0) ? wq : (mat == 1) ? wk : wv;

  v8f acc[2][4] = {};
  gemm_tile_128x128(xh + (size_t)mt * 128 * D_MODEL,
                    W  + (size_t)nt * 128 * D_MODEL,
                    SEQ, D_MODEL, &Abuf[0][0], &Bbuf[0][0], acc);

  const int lane = threadIdx.x & 31;
  const int lwid = threadIdx.x >> 5;
  const int wm = lwid & 3, wn = lwid >> 2;
  const int c  = lane & 15;
  const int hi = (lane & 16) >> 1;

  if (mat == 2) {
    // V: transposed store, 8 consecutive seq positions per register.
    #pragma unroll
    for (int im = 0; im < 2; ++im) {
      #pragma unroll
      for (int jn = 0; jn < 4; ++jn) {
        const int o     = nt * 128 + wn * 64 + jn * 16 + c;
        const int sbase = mt * 128 + wm * 32 + im * 16 + hi;
        h8 v;
        #pragma unroll
        for (int r = 0; r < 8; ++r) v[r] = (_Float16)acc[im][jn][r];
        *(h8*)(Vt + (size_t)o * SEQ + sbase) = v;
      }
    }
  } else {
    _Float16* Out = (mat == 0) ? Qh : Kh;
    #pragma unroll
    for (int im = 0; im < 2; ++im) {
      #pragma unroll
      for (int jn = 0; jn < 4; ++jn) {
        const int o  = nt * 128 + wn * 64 + jn * 16 + c;
        const int h  = o >> 6, dk = o & 63;
        #pragma unroll
        for (int r = 0; r < 8; ++r) {
          const int s = mt * 128 + wm * 32 + im * 16 + hi + r;
          Out[((size_t)h * SEQ + s) * D_K + dk] = (_Float16)acc[im][jn][r];
        }
      }
    }
  }
}

// ---------------------------------------------------------------------------
// Interleaved RoPE in place on Q and K ([h][s][64] f16), f32 math.
__global__ __launch_bounds__(256) void rope_kernel(_Float16* __restrict__ Qh,
                                                   _Float16* __restrict__ Kh) {
  const int idx = blockIdx.x * blockDim.x + threadIdx.x;
  if (idx >= N_HEADS * SEQ * (D_K / 2)) return;
  const int j = idx & 31;
  const int s = (idx >> 5) & (SEQ - 1);
  const int h = idx >> 17;

  const float inv = __powf(10000.0f, -(float)j / 32.0f);
  float sn, cs;
  __sincosf((float)s * inv, &sn, &cs);

  const size_t base = ((size_t)h * SEQ + s) * D_K + 2 * j;
  const float q0 = (float)Qh[base], q1 = (float)Qh[base + 1];
  Qh[base]     = (_Float16)(q0 * cs - q1 * sn);
  Qh[base + 1] = (_Float16)(q1 * cs + q0 * sn);
  const float k0 = (float)Kh[base], k1 = (float)Kh[base + 1];
  Kh[base]     = (_Float16)(k0 * cs - k1 * sn);
  Kh[base + 1] = (_Float16)(k1 * cs + k0 * sn);
}

// ---------------------------------------------------------------------------
// Flash-style causal attention, one wave per (head, 32-query tile).
// S = Q.K^T: 8 WMMAs / 32-key chunk (K frags shared by both query tiles).
// P.V: 8 WMMAs / chunk, V read contiguously from transposed Vt[h][64][seq].
__global__ __launch_bounds__(256) void attn_kernel(const _Float16* __restrict__ Qh,
                                                   const _Float16* __restrict__ Kh,
                                                   const _Float16* __restrict__ Vt,
                                                   _Float16* __restrict__ Oh) {
  __shared__ _Float16 Pst[8][32][32];   // per-wave 32x32 P staging (16KB)

  const int lane = threadIdx.x & 31;
  const int lwid = threadIdx.x >> 5;
  const int wid  = blockIdx.x * 8 + lwid;
  const int h    = wid >> 7;            // SEQ/32 = 128 query tiles per head
  const int qt   = wid & 127;
  const int q0   = qt * 32;

  const _Float16* Qb = Qh + ((size_t)h * SEQ + q0) * D_K;
  const _Float16* Kb = Kh + (size_t)h * SEQ * D_K;
  const _Float16* Vb = Vt + (size_t)h * D_K * SEQ;

  v16h aq[2][2];
  #pragma unroll
  for (int qi = 0; qi < 2; ++qi) {
    aq[qi][0] = load_a(Qb + (size_t)qi * 16 * D_K, D_K, 0, lane);
    aq[qi][1] = load_a(Qb + (size_t)qi * 16 * D_K, D_K, 32, lane);
  }

  v8f oacc[2][4] = {};
  float m[2][8], l[2][8];
  #pragma unroll
  for (int qi = 0; qi < 2; ++qi)
    #pragma unroll
    for (int r = 0; r < 8; ++r) { m[qi][r] = -1e30f; l[qi][r] = 0.0f; }

  const int hi = (lane & 16) >> 1;
  const int c  = lane & 15;
  const int nch = (q0 + 32 + 31) >> 5;

  for (int kc = 0; kc < nch; ++kc) {
    const int kb0 = kc * 32;

    if (kc + 1 < nch) {   // global_prefetch_b8 for next K/V chunk
      __builtin_prefetch((const void*)(Kb + (size_t)(kb0 + 32) * D_K + lane * 64), 0, 1);
      __builtin_prefetch((const void*)(Vb + (size_t)lane * SEQ + kb0 + 32), 0, 1);
      __builtin_prefetch((const void*)(Vb + (size_t)(lane + 32) * SEQ + kb0 + 32), 0, 1);
    }

    // K fragments for both 16-key subtiles (shared by both query tiles).
    v16h kf[4];
    #pragma unroll
    for (int t = 0; t < 2; ++t) {
      const _Float16* Kt = Kb + (size_t)(kb0 + t * 16) * D_K;
      kf[t * 2 + 0] = load_b(Kt, D_K, 0, lane);
      kf[t * 2 + 1] = load_b(Kt, D_K, 32, lane);
    }

    #pragma unroll
    for (int qi = 0; qi < 2; ++qi) {
      float s[2][8];
      #pragma unroll
      for (int t = 0; t < 2; ++t) {
        v8f cs = {};
        cs = __builtin_amdgcn_wmma_f32_16x16x32_f16(false, aq[qi][0], false, kf[t * 2 + 0],
                                                    (short)0, cs, false, false);
        cs = __builtin_amdgcn_wmma_f32_16x16x32_f16(false, aq[qi][1], false, kf[t * 2 + 1],
                                                    (short)0, cs, false, false);
        const int kcol = kb0 + t * 16 + c;
        #pragma unroll
        for (int r = 0; r < 8; ++r) {
          const int qrow = q0 + qi * 16 + hi + r;
          s[t][r] = (kcol > qrow) ? -1e30f : cs[r] * 0.125f;
        }
      }

      // online softmax per row (16 lanes hold each row)
      #pragma unroll
      for (int r = 0; r < 8; ++r) {
        float v = fmaxf(s[0][r], s[1][r]);
        #pragma unroll
        for (int off = 1; off < 16; off <<= 1)
          v = fmaxf(v, __shfl_xor(v, off, 32));
        const float mn   = fmaxf(m[qi][r], v);
        const float corr = __expf(m[qi][r] - mn);
        m[qi][r] = mn;
        const float p0 = __expf(s[0][r] - mn);
        const float p1 = __expf(s[1][r] - mn);
        s[0][r] = p0; s[1][r] = p1;
        float rs = p0 + p1;
        #pragma unroll
        for (int off = 1; off < 16; off <<= 1)
          rs += __shfl_xor(rs, off, 32);
        l[qi][r] = l[qi][r] * corr + rs;
        oacc[qi][0][r] *= corr; oacc[qi][1][r] *= corr;
        oacc[qi][2][r] *= corr; oacc[qi][3][r] *= corr;
      }

      // stage P (C-layout) into wave-private LDS (DS in-order per wave)
      #pragma unroll
      for (int t = 0; t < 2; ++t)
        #pragma unroll
        for (int r = 0; r < 8; ++r)
          Pst[lwid][qi * 16 + hi + r][t * 16 + c] = (_Float16)s[t][r];
    }

    // reload both P tiles in A layout
    const int pr = lane & 15;
    const int pk = (lane & 16) >> 1;
    v16h pa[2];
    #pragma unroll
    for (int qi = 0; qi < 2; ++qi) {
      const _Float16* pp = &Pst[lwid][qi * 16 + pr][0];
      pa[qi] = pack16(*(const h8*)(pp + pk), *(const h8*)(pp + pk + 16));
    }

    // V fragments from transposed Vt: two contiguous 8-key runs per lane (table k order).
    #pragma unroll
    for (int t4 = 0; t4 < 4; ++t4) {
      const v16h vf = load_b(Vb + (size_t)(t4 * 16) * SEQ, SEQ, kb0, lane);
      oacc[0][t4] = __builtin_amdgcn_wmma_f32_16x16x32_f16(false, pa[0], false, vf,
                                                           (short)0, oacc[0][t4],
                                                           false, false);
      oacc[1][t4] = __builtin_amdgcn_wmma_f32_16x16x32_f16(false, pa[1], false, vf,
                                                           (short)0, oacc[1][t4],
                                                           false, false);
    }
  }

  #pragma unroll
  for (int qi = 0; qi < 2; ++qi)
    #pragma unroll
    for (int t4 = 0; t4 < 4; ++t4)
      #pragma unroll
      for (int r = 0; r < 8; ++r) {
        const int srow = q0 + qi * 16 + hi + r;
        Oh[(size_t)srow * D_MODEL + h * D_K + t4 * 16 + c] =
            (_Float16)(oacc[qi][t4][r] / l[qi][r]);
      }
}

// ---------------------------------------------------------------------------
// Output projection: out[s,o] = sum_i O[s,i] * Wo[o,i], f32 result.
__global__ __launch_bounds__(256) void out_gemm(const _Float16* __restrict__ Oh,
                                                const _Float16* __restrict__ woh,
                                                float* __restrict__ out) {
  __shared__ _Float16 Abuf[2][128 * 32];
  __shared__ _Float16 Bbuf[2][128 * 32];

  const int mt = blockIdx.x >> 3;
  const int nt = blockIdx.x & 7;

  v8f acc[2][4] = {};
  gemm_tile_128x128(Oh  + (size_t)mt * 128 * D_MODEL,
                    woh + (size_t)nt * 128 * D_MODEL,
                    SEQ, D_MODEL, &Abuf[0][0], &Bbuf[0][0], acc);

  const int lane = threadIdx.x & 31;
  const int lwid = threadIdx.x >> 5;
  const int wm = lwid & 3, wn = lwid >> 2;
  const int c  = lane & 15;
  const int hi = (lane & 16) >> 1;
  #pragma unroll
  for (int im = 0; im < 2; ++im) {
    #pragma unroll
    for (int jn = 0; jn < 4; ++jn) {
      const int o = nt * 128 + wn * 64 + jn * 16 + c;
      #pragma unroll
      for (int r = 0; r < 8; ++r) {
        const int s = mt * 128 + wm * 32 + im * 16 + hi + r;
        out[(size_t)s * D_MODEL + o] = acc[im][jn][r];
      }
    }
  }
}

// ---------------------------------------------------------------------------
extern "C" void kernel_launch(void* const* d_in, const int* in_sizes, int n_in,
                              void* d_out, int out_size, void* d_ws, size_t ws_size,
                              hipStream_t stream) {
  (void)in_sizes; (void)n_in; (void)out_size; (void)ws_size;

  const float* x  = (const float*)d_in[0];
  const float* Wq = (const float*)d_in[1];
  const float* Wk = (const float*)d_in[2];
  const float* Wv = (const float*)d_in[3];
  const float* Wo = (const float*)d_in[4];

  const size_t NX = (size_t)SEQ * D_MODEL;
  const size_t NW = (size_t)D_MODEL * D_MODEL;

  _Float16* xh  = (_Float16*)d_ws;
  _Float16* wqh = xh  + NX;
  _Float16* wkh = wqh + NW;
  _Float16* wvh = wkh + NW;
  _Float16* woh = wvh + NW;
  _Float16* Qh  = woh + NW;       // [h][s][64]
  _Float16* Kh  = Qh  + NX;       // [h][s][64]
  _Float16* Vt  = Kh  + NX;       // [h][64][seq]  (transposed V)
  _Float16* Oh  = Vt  + NX;       // [s][1024]

  cvt_f32_f16<<<4096, 256, 0, stream>>>(x,  xh,  (int)NX);
  cvt_f32_f16<<<1024, 256, 0, stream>>>(Wq, wqh, (int)NW);
  cvt_f32_f16<<<1024, 256, 0, stream>>>(Wk, wkh, (int)NW);
  cvt_f32_f16<<<1024, 256, 0, stream>>>(Wv, wvh, (int)NW);
  cvt_f32_f16<<<1024, 256, 0, stream>>>(Wo, woh, (int)NW);

  qkv_gemm<<<768, 256, 0, stream>>>(xh, wqh, wkh, wvh, Qh, Kh, Vt);

  rope_kernel<<<(N_HEADS * SEQ * (D_K / 2)) / 256, 256, 0, stream>>>(Qh, Kh);

  // 16 heads * 128 (32-wide) query tiles = 2048 waves / 8 per block
  attn_kernel<<<256, 256, 0, stream>>>(Qh, Kh, Vt, Oh);

  out_gemm<<<256, 256, 0, stream>>>(Oh, woh, (float*)d_out);
}
